// PersistentMemoryAttention_14456859918523
// MI455X (gfx1250) — matrix operand order repaired and checked by reading, not compiled, not run on hardware
//
#include <hip/hip_runtime.h>

// ---- problem constants (fixed by reference) ----
#define Bn   2
#define Tn   2048
#define Cn   1024
#define NHn  16
#define NKVn 4
#define HDn  64
#define Mn   64
#define GCn  32
#define Sn   (Mn + Tn)          // 2112, multiple of 32
#define EPSf 1e-6f

typedef __attribute__((ext_vector_type(16))) _Float16 v16h;
typedef __attribute__((ext_vector_type(8)))  _Float16 v8h;
typedef __attribute__((ext_vector_type(8)))  float    v8f;

static __device__ __forceinline__ v16h join8(v8h lo, v8h hi) {
  return __builtin_shufflevector(lo, hi, 0,1,2,3,4,5,6,7,8,9,10,11,12,13,14,15);
}
static __device__ __forceinline__ v8f wmma16(v16h a, v16h b, v8f c) {
  return __builtin_amdgcn_wmma_f32_16x16x32_f16(false, a, false, b, (short)0, c, false, false);
}
static __device__ __forceinline__ v8f zero8() {
  v8f z = {0.f,0.f,0.f,0.f,0.f,0.f,0.f,0.f};
  return z;
}

// ---------------- f32 -> f16 convert ----------------
__global__ void cvt_f32_f16(const float* __restrict__ src, _Float16* __restrict__ dst, int n) {
  int i = blockIdx.x * blockDim.x + threadIdx.x;
  if (i < n) dst[i] = (_Float16)src[i];
}

// ---------------- WMMA GEMM, 32x32 register-blocked: C[m,n] = sum_k A[m,k]*W[n,k] ----------------
// A: (Mrows,K) f16 row-major.  W: (N,K) f16 row-major.  C: (Mrows,N) f32.
// One wave owns a 32x32 output block = 2x2 WMMA tiles; each A/B fragment feeds 2 WMMAs.
__global__ void wmma_gemm_xwt(const _Float16* __restrict__ A, const _Float16* __restrict__ W,
                              float* __restrict__ Cout, int Mrows, int N, int K) {
  const int wave = threadIdx.x >> 5;
  const int lane = threadIdx.x & 31;
  const int lq   = lane & 15;
  const int half = lane >> 4;
  const int tN   = N >> 5;                         // 32-wide column blocks
  const int tile = blockIdx.x * (blockDim.x >> 5) + wave;
  const int total = (Mrows >> 5) * tN;
  if (tile >= total) return;
  const int m0 = (tile / tN) << 5;
  const int n0 = (tile % tN) << 5;

  v8f acc00 = zero8(), acc01 = zero8(), acc10 = zero8(), acc11 = zero8();
  const _Float16* arow0 = A + (size_t)(m0 + lq) * K;
  const _Float16* arow1 = A + (size_t)(m0 + 16 + lq) * K;
  const _Float16* wrow0 = W + (size_t)(n0 + lq) * K;
  const _Float16* wrow1 = W + (size_t)(n0 + 16 + lq) * K;
  for (int k0 = 0; k0 < K; k0 += 32) {
    // A-matrix: element e -> k = k0 + 16*half + e (contiguous per half-wave)
    v16h a0 = *(const v16h*)(arow0 + k0 + 16*half);
    v16h a1 = *(const v16h*)(arow1 + k0 + 16*half);
    // B-matrix: element e -> k = k0 + 16*half + e (contiguous per half-wave)
    v16h b0 = *(const v16h*)(wrow0 + k0 + 16*half);
    v16h b1 = *(const v16h*)(wrow1 + k0 + 16*half);
    acc00 = wmma16(a0, b0, acc00);
    acc01 = wmma16(a0, b1, acc01);
    acc10 = wmma16(a1, b0, acc10);
    acc11 = wmma16(a1, b1, acc11);
  }
#pragma unroll
  for (int r = 0; r < 8; ++r) {
    const size_t row0 = (size_t)(m0 + r + 8 * half) * N;
    const size_t row1 = (size_t)(m0 + 16 + r + 8 * half) * N;
    Cout[row0 + n0 + lq]      = acc00[r];
    Cout[row0 + n0 + 16 + lq] = acc01[r];
    Cout[row1 + n0 + lq]      = acc10[r];
    Cout[row1 + n0 + 16 + lq] = acc11[r];
  }
}

// ---------------- RoPE + RMS (wave per row of HD=64) ----------------
// in: (B,T,H,HD) f32.  out: f16 at ((b*Sdim + t + tOff)*H + h)*HD
__global__ void rope_rms_kernel(const float* __restrict__ in, const float* __restrict__ cosT,
                                const float* __restrict__ sinT, _Float16* __restrict__ out,
                                int H, int Sdim, int tOff, float scale, int nrows) {
  int row  = blockIdx.x * (blockDim.x >> 5) + (threadIdx.x >> 5);
  int lane = threadIdx.x & 31;
  if (row >= nrows) return;
  int hh = row % H;
  int t  = (row / H) % Tn;
  int b  = row / (H * Tn);
  const float* p = in + (size_t)row * HDn;
  float x1 = p[lane], x2 = p[lane + 32];
  float c = cosT[t * 32 + lane], s = sinT[t * 32 + lane];
  float r1 = x1 * c - x2 * s;
  float r2 = x1 * s + x2 * c;
  float ss = r1 * r1 + r2 * r2;
#pragma unroll
  for (int m = 16; m >= 1; m >>= 1) ss += __shfl_xor(ss, m, 32);
  float rms = rsqrtf(ss * (1.0f / HDn) + EPSf) * scale;
  size_t o = (((size_t)b * Sdim + t + tOff) * H + hh) * HDn;
  out[o + lane]      = (_Float16)(r1 * rms);
  out[o + lane + 32] = (_Float16)(r2 * rms);
}

// ---------------- mem_k RMS -> k_full[s<M] ----------------
__global__ void memk_rms_kernel(const float* __restrict__ memk, _Float16* __restrict__ kfull, int nrows) {
  int row  = blockIdx.x * (blockDim.x >> 5) + (threadIdx.x >> 5);
  int lane = threadIdx.x & 31;
  if (row >= nrows) return;                 // rows = B*M*NKV
  int kv = row % NKVn;
  int m  = (row / NKVn) % Mn;
  int b  = row / (NKVn * Mn);
  const float* p = memk + ((size_t)m * NKVn + kv) * HDn;
  float x1 = p[lane], x2 = p[lane + 32];
  float ss = x1 * x1 + x2 * x2;
#pragma unroll
  for (int msk = 16; msk >= 1; msk >>= 1) ss += __shfl_xor(ss, msk, 32);
  float rms = rsqrtf(ss * (1.0f / HDn) + EPSf) * 1.2f;
  size_t o = (((size_t)b * Sn + m) * NKVn + kv) * HDn;
  kfull[o + lane]      = (_Float16)(x1 * rms);
  kfull[o + lane + 32] = (_Float16)(x2 * rms);
}

// ---------------- v fuse: v += gate*ve ; write transposed vT (B,NKV,HD,Sn) ----------------
__global__ void vfuse_kernel(const float* __restrict__ vraw, const float* __restrict__ x,
                             const float* __restrict__ ve, const float* __restrict__ Wg,
                             _Float16* __restrict__ vT, int nrows) {
  int row  = blockIdx.x * (blockDim.x >> 5) + (threadIdx.x >> 5);
  int lane = threadIdx.x & 31;
  if (row >= nrows) return;                 // rows = B*T*NKV
  int kv = row % NKVn;
  int t  = (row / NKVn) % Tn;
  int b  = row / (NKVn * Tn);
  float gp = x[((size_t)b * Tn + t) * Cn + lane] * Wg[kv * GCn + lane];
#pragma unroll
  for (int m = 16; m >= 1; m >>= 1) gp += __shfl_xor(gp, m, 32);
  float gate = 3.0f / (1.0f + __expf(-gp));
  const float* vp  = vraw + (size_t)row * HDn;
  const float* vep = ve + (((size_t)b * Tn + t) * NKVn + kv) * HDn;
  float v1 = vp[lane]      + gate * vep[lane];
  float v2 = vp[lane + 32] + gate * vep[lane + 32];
  size_t base = ((size_t)b * NKVn + kv) * HDn;
  int s = Mn + t;
  vT[(base + lane)      * Sn + s] = (_Float16)v1;
  vT[(base + lane + 32) * Sn + s] = (_Float16)v2;
}

// ---------------- mem_v * v_scale -> vT[s<M] ----------------
__global__ void memv_kernel(const float* __restrict__ memv, const float* __restrict__ vs,
                            _Float16* __restrict__ vT, int nrows) {
  int row  = blockIdx.x * (blockDim.x >> 5) + (threadIdx.x >> 5);
  int lane = threadIdx.x & 31;
  if (row >= nrows) return;                 // rows = B*M*NKV
  int kv = row % NKVn;
  int m  = (row / NKVn) % Mn;
  int b  = row / (NKVn * Mn);
  float sc = vs[0];
  const float* p = memv + ((size_t)m * NKVn + kv) * HDn;
  size_t base = ((size_t)b * NKVn + kv) * HDn;
  vT[(base + lane)      * Sn + m] = (_Float16)(p[lane] * sc);
  vT[(base + lane + 32) * Sn + m] = (_Float16)(p[lane + 32] * sc);
}

// ---------------- flash attention: one wave per (b,h,16-query tile) ----------------
// qh: (B,T,NH,HD) f16 (scaled by 1.2/8).  kf: (B,Sn,NKV,HD) f16 (scaled 1.2).
// vT: (B,NKV,HD,Sn) f16.  yh: (B,T,NH*HD) f16.
__global__ void flash_attn_kernel(const _Float16* __restrict__ qh, const _Float16* __restrict__ kf,
                                  const _Float16* __restrict__ vT, _Float16* __restrict__ yh) {
  const int wave = threadIdx.x >> 5;
  const int lane = threadIdx.x & 31;
  const int lq   = lane & 15;
  const int half = lane >> 4;
  const int T16  = Tn / 16;
  int tile = blockIdx.x * (blockDim.x >> 5) + wave;
  if (tile >= Bn * NHn * T16) return;
  const int b  = tile / (NHn * T16);
  const int h  = (tile / T16) % NHn;
  const int t0 = (tile % T16) * 16;
  const int kv = h >> 2;                    // NH/NKV = 4
  const int tg = t0 + lq;                   // this lane's query (column of S^T)

  // Q^T B-matrix chunks: e<8 -> hd = 32*chunk + 8*half + e ; e>=8 -> 32*chunk + 16 + 8*half + (e-8)
  const _Float16* qrow = qh + (((size_t)b * Tn + tg) * NHn + h) * HDn;
  v16h qb0 = join8(*(const v8h*)(qrow + 8 * half), *(const v8h*)(qrow + 16 + 8 * half));
  v16h qb1 = join8(*(const v8h*)(qrow + 32 + 8 * half), *(const v8h*)(qrow + 48 + 8 * half));

  v8f oacc[4] = { zero8(), zero8(), zero8(), zero8() };
  float run_max = -1e30f, run_sum = 0.0f;

  int send = (Mn + t0 + 16 + 31) & ~31;
  if (send > Sn) send = Sn;

  const int krow = NKVn * HDn;              // 256
  const _Float16* kbase = kf + (size_t)b * Sn * krow + (size_t)kv * HDn;
  const _Float16* vbase = vT + ((size_t)b * NKVn + kv) * HDn * Sn;

  for (int s0 = 0; s0 < send; s0 += 32) {
    // ---- scores: S^T = K * Q^T, two 16-key sub-tiles ----
    const _Float16* kr0 = kbase + (size_t)(s0 + lq) * krow;
    const _Float16* kr1 = kr0 + 16 * krow;
    __builtin_prefetch(kr0 + 32 * krow, 0, 1);   // next key tile -> global_prefetch
    v8f st0 = zero8(), st1 = zero8();
    {
      v16h a0 = join8(*(const v8h*)(kr0 + 8*half),      *(const v8h*)(kr0 + 16 + 8*half));
      v16h a1 = join8(*(const v8h*)(kr0 + 32 + 8*half), *(const v8h*)(kr0 + 48 + 8*half));
      st0 = wmma16(a0, qb0, st0);
      st0 = wmma16(a1, qb1, st0);
    }
    {
      v16h a0 = join8(*(const v8h*)(kr1 + 8*half),      *(const v8h*)(kr1 + 16 + 8*half));
      v16h a1 = join8(*(const v8h*)(kr1 + 32 + 8*half), *(const v8h*)(kr1 + 48 + 8*half));
      st1 = wmma16(a0, qb0, st1);
      st1 = wmma16(a1, qb1, st1);
    }
    // ---- causal mask + online softmax (query column = lq; lane pair (l, l^16)) ----
    float pv0[8], pv1[8];
    float mloc = -1e30f;
#pragma unroll
    for (int r = 0; r < 8; ++r) {
      int sg0 = s0 + r + 8 * half;          // key index of st0[r]
      int sg1 = sg0 + 16;                   // key index of st1[r]
      float a = ((sg0 < Mn) || (sg0 - Mn <= tg)) ? st0[r] : -1e30f;
      float c = ((sg1 < Mn) || (sg1 - Mn <= tg)) ? st1[r] : -1e30f;
      pv0[r] = a; pv1[r] = c;
      mloc = fmaxf(mloc, fmaxf(a, c));
    }
    mloc = fmaxf(mloc, __shfl_xor(mloc, 16, 32));
    float mnew = fmaxf(run_max, mloc);
    float corr = __expf(run_max - mnew);
    float ssum = 0.0f;
#pragma unroll
    for (int r = 0; r < 8; ++r) {
      pv0[r] = (pv0[r] > -1e29f) ? __expf(pv0[r] - mnew) : 0.0f;
      pv1[r] = (pv1[r] > -1e29f) ? __expf(pv1[r] - mnew) : 0.0f;
      ssum += pv0[r] + pv1[r];
    }
    ssum += __shfl_xor(ssum, 16, 32);
    run_sum = run_sum * corr + ssum;
    run_max = mnew;
#pragma unroll
    for (int cd = 0; cd < 4; ++cd)
#pragma unroll
      for (int r = 0; r < 8; ++r) oacc[cd][r] *= corr;

    // ---- assemble P^T as B-matrix (one shfl per reg pair swaps cross-half blocks) ----
    v16h pb;
#pragma unroll
    for (int r = 0; r < 8; ++r) {
      float sendv = half ? pv0[r] : pv1[r];
      float recv  = __shfl_xor(sendv, 16, 32);
      pb[r]     = (_Float16)(half ? recv   : pv0[r]);  // element r     -> key r + 16*half
      pb[8 + r] = (_Float16)(half ? pv1[r] : recv);    // element 8 + r -> key 8+r + 16*half
    }
    // ---- O^T += V^T * P^T (4 head-dim chunks of 16) ----
#pragma unroll
    for (int cd = 0; cd < 4; ++cd) {
      const _Float16* vp = vbase + (size_t)(16 * cd + lq) * Sn + s0 + 16 * half;
      v16h av = join8(*(const v8h*)vp, *(const v8h*)(vp + 8));
      oacc[cd] = wmma16(av, pb, oacc[cd]);
    }
  }

  float rinv = 1.0f / run_sum;
  size_t ybase = ((size_t)b * Tn + t0 + lq) * Cn + (size_t)h * HDn;
#pragma unroll
  for (int cd = 0; cd < 4; ++cd)
#pragma unroll
    for (int r = 0; r < 8; ++r) {
      int d = 16 * cd + r + 8 * half;
      yh[ybase + d] = (_Float16)(oacc[cd][r] * rinv);
    }
}

// ================= host side =================
extern "C" void kernel_launch(void* const* d_in, const int* in_sizes, int n_in,
                              void* d_out, int out_size, void* d_ws, size_t ws_size,
                              hipStream_t stream) {
  const float* x    = (const float*)d_in[0];
  const float* ve   = (const float*)d_in[1];
  const float* cosT = (const float*)d_in[2];
  const float* sinT = (const float*)d_in[3];
  const float* Wq   = (const float*)d_in[4];
  const float* Wk   = (const float*)d_in[5];
  const float* Wv   = (const float*)d_in[6];
  const float* Wp   = (const float*)d_in[7];
  const float* Wg   = (const float*)d_in[8];
  const float* memk = (const float*)d_in[9];
  const float* memv = (const float*)d_in[10];
  const float* vsc  = (const float*)d_in[11];
  float* out = (float*)d_out;

  char* ws = (char*)d_ws;
  size_t off = 0;
  auto carve = [&](size_t bytes) -> void* {
    void* p = ws + off;
    off = (off + bytes + 255) & ~(size_t)255;
    return p;
  };

  const size_t NX   = (size_t)Bn * Tn * Cn;            // 4,194,304
  const size_t NQ   = (size_t)Bn * Tn * NHn * HDn;     // 4,194,304
  const size_t NKV_ = (size_t)Bn * Tn * NKVn * HDn;    // 1,048,576
  const size_t NKF  = (size_t)Bn * Sn * NKVn * HDn;    // 1,081,344

  _Float16* xh    = (_Float16*)carve(NX * 2);
  _Float16* wqh   = (_Float16*)carve((size_t)Cn * Cn * 2);
  _Float16* wkh   = (_Float16*)carve((size_t)NKVn * HDn * Cn * 2);
  _Float16* wvh   = (_Float16*)carve((size_t)NKVn * HDn * Cn * 2);
  _Float16* wph   = (_Float16*)carve((size_t)Cn * Cn * 2);
  float*    qraw  = (float*)carve(NQ * 4);
  float*    kraw  = (float*)carve(NKV_ * 4);
  float*    vraw  = (float*)carve(NKV_ * 4);
  _Float16* qh    = (_Float16*)carve(NQ * 2);
  _Float16* kfull = (_Float16*)carve(NKF * 2);
  _Float16* vTh   = (_Float16*)carve(NKF * 2);
  _Float16* yh    = (_Float16*)carve(NX * 2);

  dim3 blk(128);
  auto cvt = [&](const float* s, _Float16* d, size_t n) {
    cvt_f32_f16<<<dim3((unsigned)((n + 255) / 256)), dim3(256), 0, stream>>>(s, d, (int)n);
  };
  // 1) convert activations + weights to f16
  cvt(x,  xh,  NX);
  cvt(Wq, wqh, (size_t)Cn * Cn);
  cvt(Wk, wkh, (size_t)NKVn * HDn * Cn);
  cvt(Wv, wvh, (size_t)NKVn * HDn * Cn);
  cvt(Wp, wph, (size_t)Cn * Cn);

  // 2) QKV projections (WMMA, 32x32 blocks)
  const int Mrows = Bn * Tn;                           // 4096
  {
    int tiles = (Mrows / 32) * (Cn / 32);              // 4096
    wmma_gemm_xwt<<<dim3(tiles / 4), blk, 0, stream>>>(xh, wqh, qraw, Mrows, Cn, Cn);
  }
  {
    int tiles = (Mrows / 32) * ((NKVn * HDn) / 32);    // 1024
    wmma_gemm_xwt<<<dim3(tiles / 4), blk, 0, stream>>>(xh, wkh, kraw, Mrows, NKVn * HDn, Cn);
    wmma_gemm_xwt<<<dim3(tiles / 4), blk, 0, stream>>>(xh, wvh, vraw, Mrows, NKVn * HDn, Cn);
  }

  // 3) elementwise fusions
  {
    int rows = Bn * Tn * NHn;                          // 65536
    rope_rms_kernel<<<dim3(rows / 4), blk, 0, stream>>>(qraw, cosT, sinT, qh,
                                                        NHn, Tn, 0, 1.2f / 8.0f, rows);
  }
  {
    int rows = Bn * Tn * NKVn;                         // 16384
    rope_rms_kernel<<<dim3(rows / 4), blk, 0, stream>>>(kraw, cosT, sinT, kfull,
                                                        NKVn, Sn, Mn, 1.2f, rows);
    vfuse_kernel<<<dim3(rows / 4), blk, 0, stream>>>(vraw, x, ve, Wg, vTh, rows);
  }
  {
    int rows = Bn * Mn * NKVn;                         // 512
    memk_rms_kernel<<<dim3(rows / 4), blk, 0, stream>>>(memk, kfull, rows);
    memv_kernel<<<dim3(rows / 4), blk, 0, stream>>>(memv, vsc, vTh, rows);
  }

  // 4) flash attention (WMMA)
  {
    int tiles = Bn * NHn * (Tn / 16);                  // 4096
    flash_attn_kernel<<<dim3(tiles / 4), blk, 0, stream>>>(qh, kfull, vTh, yh);
  }

  // 5) output projection (WMMA, 32x32 blocks) -> f32 d_out
  {
    int tiles = (Mrows / 32) * (Cn / 32);              // 4096
    wmma_gemm_xwt<<<dim3(tiles / 4), blk, 0, stream>>>(yh, wph, out, Mrows, Cn, Cn);
  }
  (void)in_sizes; (void)n_in; (void)out_size; (void)ws_size;
}
